// DistanceNetwork_43353399886117
// MI455X (gfx1250) — matrix operand order repaired and checked by reading, not compiled, not run on hardware
//
#include <hip/hip_runtime.h>
#include <math.h>

// Problem dims (fixed by the reference).
constexpr int NROWS = 4096;  // target rows
constexpr int MROWS = 4096;  // ss rows
constexpr int KDIM  = 1024;

// Tiling.
constexpr int BM = 128;        // block rows (target)
constexpr int BN = 128;        // block cols (ss)
constexpr int BK = 32;         // k-step (matches wmma 16x16x32)
constexpr int LDSS = 40;       // bf16 elems per LDS row (32 + 8 pad -> 80B, bank-friendly)
constexpr int NKT = KDIM / BK; // 32 k-steps

typedef __attribute__((ext_vector_type(16))) __bf16 v16bf;
typedef __attribute__((ext_vector_type(8)))  __bf16 v8bf;
typedef __attribute__((ext_vector_type(8)))  float  v8f;
typedef unsigned short ushort_t;
typedef int v4i_raw __attribute__((vector_size(16)));  // matches builtin's pointee type

// CDNA5 async global->LDS path (ASYNCcnt-tracked), with graceful fallback.
#if defined(__AMDGCN__) && __has_builtin(__builtin_amdgcn_global_load_async_to_lds_b128)
#define HAVE_ASYNC 1
#else
#define HAVE_ASYNC 0
#endif

__device__ inline unsigned short f2bf_rne(float f) {
    unsigned u = __float_as_uint(f);
    u += 0x7fffu + ((u >> 16) & 1u);
    return (unsigned short)(u >> 16);
}
__device__ inline float bf2f(unsigned short h) {
    return __uint_as_float(((unsigned)h) << 16);
}
__device__ inline uint4 pack8(const unsigned short* p) {
    return make_uint4((unsigned)p[0] | ((unsigned)p[1] << 16),
                      (unsigned)p[2] | ((unsigned)p[3] << 16),
                      (unsigned)p[4] | ((unsigned)p[5] << 16),
                      (unsigned)p[6] | ((unsigned)p[7] << 16));
}

#if HAVE_ASYNC
__device__ inline void async_cp16(const void* g, void* l) {
    // ROCm 7.2 signature: (v4i addrspace(1)* src, v4i addrspace(3)* dst, imm offset, imm cpol)
    __builtin_amdgcn_global_load_async_to_lds_b128(
        (v4i_raw __attribute__((address_space(1)))*)(g),
        (v4i_raw __attribute__((address_space(3)))*)(l), 0, 0);
}
#endif

__device__ inline void wait_async0() {
#if defined(__AMDGCN__)
#if __has_builtin(__builtin_amdgcn_s_wait_asynccnt)
    __builtin_amdgcn_s_wait_asynccnt(0);
#else
    asm volatile("s_wait_asynccnt 0" ::: "memory");
#endif
#endif
}

// Load 16 contiguous fp32, split into hi/lo bf16, store 32B each into LDS (fallback path).
__device__ inline void load_split16(const float* __restrict__ g,
                                    unsigned short* shi, unsigned short* slo) {
    unsigned short hi[16], lo[16];
    const float4* g4 = (const float4*)g;
#pragma unroll
    for (int i = 0; i < 4; ++i) {
        float4 v = g4[i];
        float xs[4] = {v.x, v.y, v.z, v.w};
#pragma unroll
        for (int j = 0; j < 4; ++j) {
            float x = xs[j];
            unsigned short hb = f2bf_rne(x);
            hi[i * 4 + j] = hb;
            lo[i * 4 + j] = f2bf_rne(x - bf2f(hb));
        }
    }
    *(uint4*)(shi)     = pack8(hi);
    *(uint4*)(shi + 8) = pack8(hi + 8);
    *(uint4*)(slo)     = pack8(lo);
    *(uint4*)(slo + 8) = pack8(lo + 8);
}

// Assemble a v16bf fragment from two 16B LDS chunks.
__device__ inline v16bf frag16(const unsigned short* p0, const unsigned short* p1) {
    v8bf a = *(const v8bf*)p0;
    v8bf b = *(const v8bf*)p1;
    return __builtin_shufflevector(a, b, 0, 1, 2, 3, 4, 5, 6, 7,
                                   8, 9, 10, 11, 12, 13, 14, 15);
}

__device__ inline v8f wmma_bf16(v16bf a, v16bf b, v8f c) {
    return __builtin_amdgcn_wmma_f32_16x16x32_bf16(false, a, false, b,
                                                   (short)0, c, false, false);
}

// ---------------- Pass 1a: row norms of both matrices ----------------
__global__ __launch_bounds__(256) void rownorms_kernel(const float* __restrict__ tgt,
                                                       const float* __restrict__ ss,
                                                       float* __restrict__ norms) {
    const int wid  = threadIdx.x >> 5;
    const int lane = threadIdx.x & 31;
    const int row  = blockIdx.x * 8 + wid;
    const float* src = blockIdx.y ? ss : tgt;
    const float4* p = (const float4*)(src + (size_t)row * KDIM);
    float s = 0.0f;
#pragma unroll
    for (int i = 0; i < 8; ++i) {
        float4 v = p[i * 32 + lane];
        s += v.x * v.x + v.y * v.y + v.z * v.z + v.w * v.w;
    }
#pragma unroll
    for (int off = 16; off > 0; off >>= 1) s += __shfl_xor(s, off, 32);
    if (lane == 0) norms[blockIdx.y * NROWS + row] = sqrtf(s);
}

// ---------------- Pass 1b: one-shot fp32 -> (hi, lo) bf16 split ----------------
__global__ __launch_bounds__(256) void convert_kernel(const float* __restrict__ tgt,
                                                      const float* __restrict__ ss,
                                                      ushort_t* __restrict__ hiA,
                                                      ushort_t* __restrict__ loA,
                                                      ushort_t* __restrict__ hiB,
                                                      ushort_t* __restrict__ loB) {
    const size_t idx = ((size_t)blockIdx.x * 256 + threadIdx.x) * 4;
    const float* src = blockIdx.y ? ss : tgt;
    ushort_t* hi = blockIdx.y ? hiB : hiA;
    ushort_t* lo = blockIdx.y ? loB : loA;
    float4 v = *(const float4*)(src + idx);
    float xs[4] = {v.x, v.y, v.z, v.w};
    ushort_t h[4], l[4];
#pragma unroll
    for (int i = 0; i < 4; ++i) {
        h[i] = f2bf_rne(xs[i]);
        l[i] = f2bf_rne(xs[i] - bf2f(h[i]));
    }
    *(uint2*)(hi + idx) = make_uint2((unsigned)h[0] | ((unsigned)h[1] << 16),
                                     (unsigned)h[2] | ((unsigned)h[3] << 16));
    *(uint2*)(lo + idx) = make_uint2((unsigned)l[0] | ((unsigned)l[1] << 16),
                                     (unsigned)l[2] | ((unsigned)l[3] << 16));
}

// ---------------- Pass 2: bf16x3 WMMA GEMM + cosine normalization ----------------
// PRECONV=true : operands are pre-split bf16 arrays; fill = async copy to LDS, double-buffered.
// PRECONV=false: operands are fp32; fill converts in-kernel (scratch-starved fallback).
template <bool PRECONV>
__global__ __launch_bounds__(256) void cosgemm_kernel(const float* __restrict__ A,
                                                      const float* __restrict__ B,
                                                      const ushort_t* __restrict__ gAhi,
                                                      const ushort_t* __restrict__ gAlo,
                                                      const ushort_t* __restrict__ gBhi,
                                                      const ushort_t* __restrict__ gBlo,
                                                      const float* __restrict__ norms,
                                                      float* __restrict__ out) {
    __shared__ unsigned short sAhi[2][BM * LDSS];
    __shared__ unsigned short sAlo[2][BM * LDSS];
    __shared__ unsigned short sBhi[2][BN * LDSS];
    __shared__ unsigned short sBlo[2][BN * LDSS];

    const int R0 = blockIdx.y * BM;
    const int C0 = blockIdx.x * BN;
    const int t  = threadIdx.x;

    // Fill-phase roles.
    const int fr = t >> 1;   // fallback: row, 2 threads per row
    const int fh = t & 1;    // fallback: which 16-wide half of the k-tile

    // Compute-phase roles: 8 waves in a 4x2 grid; each wave owns 32x64 = 2x4 tiles.
    const int lane = t & 31;
    const int wid  = t >> 5;
    const int wr   = wid >> 1;  // 0..3
    const int wc   = wid & 1;   // 0..1
    const int lr   = lane & 15;
    const bool up  = lane >= 16;

    const v8f vzero = {0.f, 0.f, 0.f, 0.f, 0.f, 0.f, 0.f, 0.f};
    v8f acc[2][4];
#pragma unroll
    for (int i = 0; i < 2; ++i)
#pragma unroll
        for (int j = 0; j < 4; ++j) acc[i][j] = vzero;

    auto fill = [&](int buf, int kt) {
        if constexpr (PRECONV) {
            // Copy 4 x (128 rows x 32 bf16) tiles; each thread moves 2 x 16B per array.
#pragma unroll
            for (int j = 0; j < 2; ++j) {
                const int c   = t + 256 * j;     // 0..511 chunk id
                const int row = c >> 2;          // 0..127
                const int off = (c & 3) * 8;     // 0,8,16,24 (elems)
                const size_t ga = (size_t)(R0 + row) * KDIM + kt + off;
                const size_t gb = (size_t)(C0 + row) * KDIM + kt + off;
                const int l = row * LDSS + off;
#if HAVE_ASYNC
                async_cp16(gAhi + ga, &sAhi[buf][l]);
                async_cp16(gAlo + ga, &sAlo[buf][l]);
                async_cp16(gBhi + gb, &sBhi[buf][l]);
                async_cp16(gBlo + gb, &sBlo[buf][l]);
#else
                *(uint4*)&sAhi[buf][l] = *(const uint4*)(gAhi + ga);
                *(uint4*)&sAlo[buf][l] = *(const uint4*)(gAlo + ga);
                *(uint4*)&sBhi[buf][l] = *(const uint4*)(gBhi + gb);
                *(uint4*)&sBlo[buf][l] = *(const uint4*)(gBlo + gb);
#endif
            }
        } else {
            const float* ga = A + (size_t)(R0 + fr) * KDIM + kt + fh * 16;
            load_split16(ga, &sAhi[buf][fr * LDSS + fh * 16],
                             &sAlo[buf][fr * LDSS + fh * 16]);
            const float* gb = B + (size_t)(C0 + fr) * KDIM + kt + fh * 16;
            load_split16(gb, &sBhi[buf][fr * LDSS + fh * 16],
                             &sBlo[buf][fr * LDSS + fh * 16]);
        }
    };

    fill(0, 0);  // prologue: stage tile 0

    for (int it = 0; it < NKT; ++it) {
        if constexpr (PRECONV) wait_async0();  // this wave's async fills done
        __syncthreads();                       // all waves: buffer (it&1) published
        if (it + 1 < NKT) fill((it + 1) & 1, (it + 1) * BK);  // overlap next fill
        const int buf = it & 1;

        // A fragments (16-bit A layout): lanes 0-15 hold K 0-7 & 16-23; 16-31 hold K 8-15 & 24-31.
        v16bf aH[2], aL[2];
        const int ac = up ? 8 : 0;
#pragma unroll
        for (int tm = 0; tm < 2; ++tm) {
            const int base = (wr * 32 + tm * 16 + lr) * LDSS;
            aH[tm] = frag16(&sAhi[buf][base + ac], &sAhi[buf][base + ac + 16]);
            aL[tm] = frag16(&sAlo[buf][base + ac], &sAlo[buf][base + ac + 16]);
        }
        // B fragments (32x16): lane n = column = row of S; lanes 0-15: K 0-7 & 16-23, 16-31: K 8-15 & 24-31.
        v16bf bH[4], bL[4];
        const int bc = up ? 8 : 0;
#pragma unroll
        for (int tb = 0; tb < 4; ++tb) {
            const int base = (wc * 64 + tb * 16 + lr) * LDSS + bc;
            bH[tb] = frag16(&sBhi[buf][base], &sBhi[buf][base + 16]);
            bL[tb] = frag16(&sBlo[buf][base], &sBlo[buf][base + 16]);
        }

        // bf16x3: hi*hi + hi*lo + lo*hi (dropped lo*lo ~2^-16 rel).
#pragma unroll
        for (int tm = 0; tm < 2; ++tm) {
#pragma unroll
            for (int tb = 0; tb < 4; ++tb) {
                acc[tm][tb] = wmma_bf16(aH[tm], bH[tb], acc[tm][tb]);
                acc[tm][tb] = wmma_bf16(aH[tm], bL[tb], acc[tm][tb]);
                acc[tm][tb] = wmma_bf16(aL[tm], bH[tb], acc[tm][tb]);
            }
        }
    }

    // Epilogue: cosine normalization + store.
    // C/D layout: VGPR v, lanes 0-15 -> M=v, lanes 16-31 -> M=v+8; N = lane%16.
    const float* tn = norms;
    const float* sn = norms + NROWS;
#pragma unroll
    for (int tm = 0; tm < 2; ++tm) {
        const int rowBase = R0 + wr * 32 + tm * 16 + (up ? 8 : 0);
#pragma unroll
        for (int tb = 0; tb < 4; ++tb) {
            const int col = C0 + wc * 64 + tb * 16 + lr;
            const float snv = sn[col];
#pragma unroll
            for (int v = 0; v < 8; ++v) {
                const int row = rowBase + v;
                const float denom = fmaxf(tn[row] * snv, 1e-8f);
                out[(size_t)row * MROWS + col] = acc[tm][tb][v] / denom;
            }
        }
    }
}

extern "C" void kernel_launch(void* const* d_in, const int* in_sizes, int n_in,
                              void* d_out, int out_size, void* d_ws, size_t ws_size,
                              hipStream_t stream) {
    (void)in_sizes; (void)n_in; (void)out_size;
    const float* target = (const float*)d_in[0];
    const float* ss     = (const float*)d_in[1];
    float* out   = (float*)d_out;
    float* norms = (float*)d_ws;  // 2*4096 floats = 32 KB, always present

    const size_t ARR      = (size_t)NROWS * KDIM;            // elems per matrix
    const size_t NORMS_B  = 32 * 1024;
    const size_t NEED     = NORMS_B + 4 * ARR * sizeof(ushort_t);  // +32 MB of bf16 splits

    rownorms_kernel<<<dim3(NROWS / 8, 2, 1), 256, 0, stream>>>(target, ss, norms);

    if (ws_size >= NEED) {
        ushort_t* base = (ushort_t*)((char*)d_ws + NORMS_B);
        ushort_t* Ahi = base;
        ushort_t* Alo = base + ARR;
        ushort_t* Bhi = base + 2 * ARR;
        ushort_t* Blo = base + 3 * ARR;
        convert_kernel<<<dim3((unsigned)(ARR / 1024), 2, 1), 256, 0, stream>>>(
            target, ss, Ahi, Alo, Bhi, Blo);
        cosgemm_kernel<true><<<dim3(MROWS / BN, NROWS / BM, 1), 256, 0, stream>>>(
            target, ss, Ahi, Alo, Bhi, Blo, norms, out);
    } else {
        cosgemm_kernel<false><<<dim3(MROWS / BN, NROWS / BM, 1), 256, 0, stream>>>(
            target, ss, nullptr, nullptr, nullptr, nullptr, norms, out);
    }
}
